// SDPAWithSink_87110526697960
// MI455X (gfx1250) — matrix operand order repaired and checked by reading, not compiled, not run on hardware
//
#include <hip/hip_runtime.h>

typedef __attribute__((ext_vector_type(16))) _Float16 v16h;
typedef __attribute__((ext_vector_type(8)))  _Float16 v8h;
typedef __attribute__((ext_vector_type(2)))  _Float16 v2h;
typedef __attribute__((ext_vector_type(8)))  float    v8f;
typedef __attribute__((ext_vector_type(4)))  float    v4f;

#define BQ     128      // q rows per block (8 waves x 16)
#define WQ     16       // q rows per wave
#define KB     64       // keys per iteration
#define DH     64       // head dim
#define LSEQ   2048
#define NH     16
#define PITCH  72       // halves per LDS row (144 B: 16B-aligned, stride 36 banks)

__device__ __forceinline__ float swz16(float x) {
  // SWAPX16: lane <-> lane^16 (pairs holding the same q row)
  return __int_as_float(__builtin_amdgcn_ds_swizzle(__float_as_int(x), 0x401F));
}

// Cooperative stage of one 64-key chunk:
//   vDst[dim][key] = f16 V^T  (transposed; paired keys pack into dword stores)
//   kDst[key][dim] = f16 K    (row-major; 16B-vector stores)
__device__ __forceinline__ void stage_chunk(const float* __restrict__ vP,
                                            const float* __restrict__ kP,
                                            int kb, _Float16* vDst, _Float16* kDst,
                                            int tid)
{
  {
    const int kp = tid & 31;                    // key pair -> keys 2kp, 2kp+1
    const int dg = tid >> 5;                    // 8 dims per group
    const float* r0 = vP + (size_t)(kb + 2 * kp) * DH + dg * 8;
    const float* r1 = r0 + DH;
    v4f a0 = ((const v4f*)r0)[0], a1 = ((const v4f*)r0)[1];
    v4f b0 = ((const v4f*)r1)[0], b1 = ((const v4f*)r1)[1];
    #pragma unroll
    for (int e = 0; e < 4; ++e) {
      v2h p0 = { (_Float16)a0[e], (_Float16)b0[e] };
      v2h p1 = { (_Float16)a1[e], (_Float16)b1[e] };
      *(v2h*)(vDst + (dg * 8 + e)     * PITCH + 2 * kp) = p0;
      *(v2h*)(vDst + (dg * 8 + 4 + e) * PITCH + 2 * kp) = p1;
    }
  }
  {
    const int key = tid & 63;
    const int dg  = tid >> 6;                   // 16 dims per group
    const float* r = kP + (size_t)(kb + key) * DH + dg * 16;
    v4f f0 = ((const v4f*)r)[0], f1 = ((const v4f*)r)[1];
    v4f f2 = ((const v4f*)r)[2], f3 = ((const v4f*)r)[3];
    v8h h0, h1;
    #pragma unroll
    for (int e = 0; e < 4; ++e) {
      h0[e]     = (_Float16)f0[e];
      h0[4 + e] = (_Float16)f1[e];
      h1[e]     = (_Float16)f2[e];
      h1[4 + e] = (_Float16)f3[e];
    }
    *(v8h*)(kDst + key * PITCH + dg * 16)     = h0;
    *(v8h*)(kDst + key * PITCH + dg * 16 + 8) = h1;
  }
}

__global__ __launch_bounds__(256, 2)
void sdpa_sink_kernel(const float* __restrict__ Q,
                      const float* __restrict__ K,
                      const float* __restrict__ V,
                      const float* __restrict__ SINK,
                      float* __restrict__ O)
{
  __shared__ __align__(16) _Float16 vT[2][DH * PITCH];  // double-buffered V^T
  __shared__ __align__(16) _Float16 kT[2][KB * PITCH];  // double-buffered K rows
  __shared__ __align__(16) float    red[8 * 16];        // per-wave row broadcast

  const int tid    = threadIdx.x;
  const int wave   = tid >> 5;
  const int lane   = tid & 31;
  const int laneLo = lane & 15;
  const int laneHi = lane >> 4;

  const int qTiles = LSEQ / BQ;
  const int bh     = blockIdx.x / qTiles;       // b*H + h
  const int qb     = blockIdx.x % qTiles;
  const int head   = bh % NH;
  const int qBase  = qb * BQ;
  const int q0     = qBase + wave * WQ;

  const size_t planeOff = (size_t)bh * LSEQ * DH;
  const float* qP = Q + planeOff;
  const float* kP = K + planeOff;
  const float* vP = V + planeOff;
  float*       oP = O + planeOff;

  // ---- Q tile in B-matrix layout (two K=32-dim chunks), scale folded in ----
  const float QSCALE = 0.125f * 1.44269504088896341f;  // 1/sqrt(D) * log2(e)
  v16h qB[2];
  {
    const float* qRow = qP + (size_t)(q0 + laneLo) * DH;
    #pragma unroll
    for (int c = 0; c < 2; ++c) {
      const v4f* p = (const v4f*)(qRow + c * 32 + laneHi * 16);
      v4f f0 = p[0], f1 = p[1], f2 = p[2], f3 = p[3];
      #pragma unroll
      for (int e = 0; e < 4; ++e) {
        qB[c][e]      = (_Float16)(f0[e] * QSCALE);
        qB[c][4 + e]  = (_Float16)(f1[e] * QSCALE);
        qB[c][8 + e]  = (_Float16)(f2[e] * QSCALE);
        qB[c][12 + e] = (_Float16)(f3[e] * QSCALE);
      }
    }
  }

  // ---- softmax state seeded by the sink (sink V = 0) ----
  float m = SINK[head] * 1.44269504088896341f;   // exp2 domain
  float l = 1.0f;                                // sink term exp2(m - m) = 1
  v8f acc[4] = {};

  const int keyEnd = qBase + BQ;                 // uniform over the block

  // ---- software pipeline: stage chunk 0, then 1 barrier per iteration ----
  stage_chunk(vP, kP, 0, vT[0], kT[0], tid);

  int buf = 0;
  for (int kb = 0; kb < keyEnd; kb += KB, buf ^= 1) {
    __syncthreads();                             // buf ready; buf^1 free to restage
    if (kb + KB < keyEnd)
      stage_chunk(vP, kP, kb + KB, vT[buf ^ 1], kT[buf ^ 1], tid);

    if (kb >= q0 + WQ) continue;                 // wave-uniform skip past causal range

    const _Float16* kTb = kT[buf];
    const _Float16* vTb = vT[buf];

    // ---- S^T = K * Q^T : four 16-key tiles, K-dim 64 split in two WMMAs ----
    v8f s[4];
    #pragma unroll
    for (int t = 0; t < 4; ++t) {
      const _Float16* kRow = kTb + (t * 16 + laneLo) * PITCH;
      v8f sAcc = {};
      #pragma unroll
      for (int c = 0; c < 2; ++c) {
        v8h lo = *(const v8h*)(kRow + c * 32 + laneHi * 16);
        v8h hi = *(const v8h*)(kRow + c * 32 + laneHi * 16 + 8);
        v16h aK = __builtin_shufflevector(lo, hi, 0, 1, 2, 3, 4, 5, 6, 7,
                                          8, 9, 10, 11, 12, 13, 14, 15);
        sAcc = __builtin_amdgcn_wmma_f32_16x16x32_f16(false, aK, false, qB[c],
                                                      (short)0, sAcc, false, false);
      }
      s[t] = sAcc;
    }

    // ---- causal mask (only blocks touching the diagonal) ----
    if (kb + KB - 1 > q0) {
      const int limit = q0 + laneLo - kb;        // key offset must be <= limit
      #pragma unroll
      for (int t = 0; t < 4; ++t)
        #pragma unroll
        for (int v = 0; v < 8; ++v) {
          const int kOff = t * 16 + laneHi * 8 + v;
          s[t][v] = (kOff <= limit) ? s[t][v] : -1e30f;
        }
    }

    // ---- online softmax over 64 keys ----
    float bm = -1e30f;
    #pragma unroll
    for (int t = 0; t < 4; ++t)
      #pragma unroll
      for (int v = 0; v < 8; ++v) bm = fmaxf(bm, s[t][v]);
    bm = fmaxf(bm, swz16(bm));
    const float mNew  = fmaxf(m, bm);
    const float alpha = exp2f(m - mNew);
    m = mNew;

    // P lands directly in A-matrix layout: pA[g] covers keys g*32 .. g*32+31
    v16h pA[2];
    float ps = 0.0f;
    #pragma unroll
    for (int g = 0; g < 2; ++g)
      #pragma unroll
      for (int half = 0; half < 2; ++half)
        #pragma unroll
        for (int v = 0; v < 8; ++v) {
          const float p = exp2f(s[2 * g + half][v] - mNew);
          ps += p;
          pA[g][half * 8 + v] = (_Float16)p;
        }
    ps += swz16(ps);
    l = l * alpha + ps;

    // ---- broadcast alpha per q-row (lane-indexed -> vgpr-indexed), rescale O ----
    if (lane < 16) red[wave * 16 + lane] = alpha;
    __builtin_amdgcn_wave_barrier();
    v4f a0 = ((const v4f*)(red + wave * 16 + laneHi * 8))[0];
    v4f a1 = ((const v4f*)(red + wave * 16 + laneHi * 8))[1];
    #pragma unroll
    for (int t = 0; t < 4; ++t)
      #pragma unroll
      for (int v = 0; v < 8; ++v)
        acc[t][v] *= (v < 4) ? a0[v & 3] : a1[v & 3];

    // ---- O += P * V : 4 dim tiles x 2 key halves (B from LDS V^T) ----
    #pragma unroll
    for (int t = 0; t < 4; ++t) {
      const _Float16* rowp = vTb + (t * 16 + laneLo) * PITCH + laneHi * 8;
      v16h vB0 = __builtin_shufflevector(*(const v8h*)(rowp), *(const v8h*)(rowp + 16),
                                         0, 1, 2, 3, 4, 5, 6, 7, 8, 9, 10, 11, 12, 13, 14, 15);           // keys kb+0  .. kb+31
      v16h vB1 = __builtin_shufflevector(*(const v8h*)(rowp + 32), *(const v8h*)(rowp + 48),
                                         0, 1, 2, 3, 4, 5, 6, 7, 8, 9, 10, 11, 12, 13, 14, 15);      // keys kb+32 .. kb+63
      acc[t] = __builtin_amdgcn_wmma_f32_16x16x32_f16(false, pA[0], false, vB0,
                                                      (short)0, acc[t], false, false);
      acc[t] = __builtin_amdgcn_wmma_f32_16x16x32_f16(false, pA[1], false, vB1,
                                                      (short)0, acc[t], false, false);
    }
  }

  // ---- finalize: 1/l per q-row, store fp32 output ----
  const float rl = 1.0f / l;
  if (lane < 16) red[wave * 16 + lane] = rl;
  __builtin_amdgcn_wave_barrier();
  v4f r0 = ((const v4f*)(red + wave * 16 + laneHi * 8))[0];
  v4f r1 = ((const v4f*)(red + wave * 16 + laneHi * 8))[1];

  #pragma unroll
  for (int t = 0; t < 4; ++t)
    #pragma unroll
    for (int v = 0; v < 8; ++v) {
      const int   row   = q0 + laneHi * 8 + v;
      const float denom = (v < 4) ? r0[v & 3] : r1[v & 3];
      oP[(size_t)row * DH + t * 16 + laneLo] = acc[t][v] * denom;
    }
}

extern "C" void kernel_launch(void* const* d_in, const int* in_sizes, int n_in,
                              void* d_out, int out_size, void* d_ws, size_t ws_size,
                              hipStream_t stream) {
  const float* q    = (const float*)d_in[0];
  const float* k    = (const float*)d_in[1];
  const float* v    = (const float*)d_in[2];
  const float* sink = (const float*)d_in[3];
  float* out = (float*)d_out;
  const int B = 4, H = 16;
  dim3 grid(B * H * (LSEQ / BQ));
  sdpa_sink_kernel<<<grid, 256, 0, stream>>>(q, k, v, sink, out);
}
